// DecoderBlock_70703751626805
// MI455X (gfx1250) — compile-verified, matrix operand order checked by reading, not run on hardware
//
#include <hip/hip_runtime.h>
#include <hip/hip_bf16.h>
#include <math.h>

// MI455X / gfx1250, wave32. bf16 WMMA (16x16x32) for all GEMM-shaped work,
// f32 accumulate. Attention / softmax / GRU gates stay f32 VALU (TRANS-heavy).
// B operands (static weights) pre-packed once into WMMA fragment-linear chunks;
// per-step GEMMs (M=64, latency-bound) use split-K with consumer-side partial
// reduction (no atomics, graph/replay deterministic).

typedef __attribute__((ext_vector_type(16))) __bf16 v16bf;
typedef __attribute__((ext_vector_type(8)))  float  v8f;

#define B_    64
#define S_    128
#define D_    512
#define T_    64
#define KGIN  544    // D+3 padded up to a multiple of 32 (WMMA K granularity)
#define N3D   1536   // 3*D (GRU gate width)
#define KS_Q  4      // split-K for q GEMM   (512/32 = 16 slabs -> 4 each)
#define KS_GI 5      // split-K for gi GEMM  (544/32 = 17 slabs -> 4/4/3/3/3)
#define KS_GH 4      // split-K for gh GEMM  (16 slabs -> 4 each)

// ---------------------------------------------------------------- conversions
__global__ void cvt_f32_to_bf16(const float* __restrict__ src,
                                __bf16* __restrict__ dst, int n) {
  int i = blockIdx.x * blockDim.x + threadIdx.x;
  if (i < n) dst[i] = (__bf16)src[i];
}

// Pack a B operand into WMMA fragment-linear chunks.
// Chunk(kt, nt) lives at ((kt * N/16) + nt) * 512 elements; inside a chunk,
// lane l owns 16 contiguous elements (32 bytes) in ISA 7.12.2 B-fragment order:
//   element e of lane l = B[kt*32 + kmap][nt*16 + (l&15)],
//   kmap = (l>>4)*8 + (e&7) + (e>>3)*16.
// kt-major chunk order => one 64-col block k-step is a contiguous 4 KB slab.
// mode 0: src is [K x N] f32 row-major (logical B directly).
// mode 1: src is [N x k_cols] f32 row-major; logical B = src^T; zero-pad k>=k_cols.
__global__ void pack_b_frag(const float* __restrict__ src, __bf16* __restrict__ dst,
                            int N, int Kpad, int k_cols, int mode) {
  int idx = blockIdx.x * blockDim.x + threadIdx.x;
  int total = N * Kpad;
  if (idx >= total) return;
  int chunk = idx >> 9, w = idx & 511;
  int lane = w >> 4, e = w & 15;
  int NT = N >> 4;
  int kt = chunk / NT, nt = chunk - kt * NT;
  int n = nt * 16 + (lane & 15);
  int k = kt * 32 + ((lane >> 4) << 3) + (e & 7) + ((e >> 3) << 4);
  float v = 0.f;
  if (k < k_cols)
    v = (mode == 0) ? src[(size_t)k * N + n] : src[(size_t)n * k_cols + k];
  dst[idx] = (__bf16)v;
}

__global__ void init_state(const float* __restrict__ e_last,
                           float* __restrict__ h, __bf16* __restrict__ hb,
                           float* __restrict__ x) {
  int i = blockIdx.x * blockDim.x + threadIdx.x;
  if (i < B_ * D_) { float v = e_last[i]; h[i] = v; hb[i] = (__bf16)v; }
  if (i < B_ * 3)  x[i] = 0.0f;
}

// ---------------------------------------------------------------- WMMA GEMM
// Split-K GEMM: partial_z[M,N] = A[M, k-slabs z, z+KS, ...] @ B + (z==0: bias).
// Sum over z (done by the consumer) gives A@B + bias.
// A bf16 row-major; Bp fragment-packed bf16 (pack_b_frag); Cpart f32, KS planes.
// grid = (N/64, M/64, KS), block = 128 (4 waves). Block tile 64x64; wave w owns
// rows [w*16, w*16+16) x 64 cols -> 4 accumulators; one A fragment (2x global
// b128) feeds 4 WMMAs per k-step; packed-B slab staged in LDS via pure b128.
// K % 32 == 0, M % 64 == 0, N % 64 == 0.
__global__ void __launch_bounds__(128)
gemm_wmma_bf16(const __bf16* __restrict__ A, const __bf16* __restrict__ Bp,
               const float* __restrict__ bias, float* __restrict__ Cpart,
               int M, int N, int K, int KS) {
  __shared__ __align__(16) __bf16 ldsBp[4 * 512];   // 4 KB slab
  const int tid   = threadIdx.x;
  const int wave  = tid >> 5;
  const int lane  = tid & 31;
  const int nBase = blockIdx.x * 64;
  const int mBase = blockIdx.y * 64 + wave * 16;
  const int split = blockIdx.z;
  const int rc    = lane & 15;          // A-row / D-col for this lane
  const int half8 = (lane >> 4) * 8;    // K sub-offset select per lane half
  const int NT    = N >> 4;
  float* C = Cpart + (size_t)split * M * N;

  v8f acc[4] = {{}, {}, {}, {}};
  for (int kt = split; kt * 32 < K; kt += KS) {
    const int k0 = kt * 32;
    // Stage contiguous 4 KB packed-B slab: 256 uint4, 2 per thread, coalesced.
    const uint4* src4 =
        (const uint4*)(Bp + ((size_t)kt * NT + (nBase >> 4)) * 512);
    uint4* dst4 = (uint4*)ldsBp;
    dst4[tid]       = src4[tid];
    dst4[tid + 128] = src4[tid + 128];
    if ((kt + KS) * 32 < K)  // warm this split's next slab
      __builtin_prefetch(
          Bp + ((size_t)(kt + KS) * NT + (nBase >> 4)) * 512, 0, 0);
    __syncthreads();

    // A fragment per ISA 7.12.2: lane L holds row (L&15);
    // elems 0..7 = K{half8..half8+7}, elems 8..15 = K{16+half8..}.
    const __bf16* arow = A + (size_t)(mBase + rc) * K + k0;
    union { v16bf v; uint4 q[2]; } af;
    af.q[0] = *(const uint4*)(arow + half8);
    af.q[1] = *(const uint4*)(arow + 16 + half8);

#pragma unroll
    for (int nt = 0; nt < 4; ++nt) {
      const __bf16* bfrag = ldsBp + nt * 512 + lane * 16;  // 32 contiguous bytes
      union { v16bf v; uint4 q[2]; } bfr;
      bfr.q[0] = *(const uint4*)(bfrag);       // ds_load_b128
      bfr.q[1] = *(const uint4*)(bfrag + 8);   // ds_load_b128
      acc[nt] = __builtin_amdgcn_wmma_f32_16x16x32_bf16(
          false, af.v, false, bfr.v, (short)0, acc[nt], false, false);
    }
    __syncthreads();
  }

  // C/D layout: VGPR i -> M = i (lanes 0-15) or 8+i (lanes 16-31), N = lane&15
#pragma unroll
  for (int nt = 0; nt < 4; ++nt) {
    const int n  = nBase + nt * 16 + rc;
    const float bv = (bias && split == 0) ? bias[n] : 0.0f;
#pragma unroll
    for (int i = 0; i < 8; ++i)
      C[(size_t)(mBase + half8 + i) * N + n] = acc[nt][i] + bv;
  }
}

// ---------------------------------------------------------------- attention
// One block per batch row b, 128 threads (thread == encoder position s).
// Sums the KS_Q split-K partial planes of q while staging it into LDS, then
// scores -> softmax -> ctx -> pack bf16 GRU input gin = [ctx | x | 0-pad].
__global__ void __launch_bounds__(128)
attn_step(const float* __restrict__ Uk, const float* __restrict__ qpart,
          const float* __restrict__ e_all, const float* __restrict__ Va_w,
          const float* __restrict__ Va_b, const float* __restrict__ x,
          float* __restrict__ cross_attn, __bf16* __restrict__ gin, int t) {
  __shared__ float qs[D_];
  __shared__ float vas[D_];
  __shared__ float sc[S_];
  __shared__ float red[128];
  const int b = blockIdx.x, s = threadIdx.x;

  // stage q row (summing split-K partials) + Va into LDS once
  for (int d = s; d < D_; d += 128) {
    float v = 0.f;
#pragma unroll
    for (int p = 0; p < KS_Q; ++p)
      v += qpart[(size_t)p * B_ * D_ + (size_t)b * D_ + d];
    qs[d] = v;
    vas[d] = Va_w[d];
  }
  __syncthreads();

  const float4* uk4 = (const float4*)(Uk + ((size_t)b * S_ + s) * D_);
  float a = 0.f;
#pragma unroll 4
  for (int d4 = 0; d4 < D_ / 4; ++d4) {
    const float4 u = uk4[d4];
    const int d = d4 * 4;
    a += tanhf(qs[d + 0] + u.x) * vas[d + 0];
    a += tanhf(qs[d + 1] + u.y) * vas[d + 1];
    a += tanhf(qs[d + 2] + u.z) * vas[d + 2];
    a += tanhf(qs[d + 3] + u.w) * vas[d + 3];
  }
  a += Va_b[0];

  // softmax over S (two LDS tree reductions)
  red[s] = a; __syncthreads();
  for (int off = 64; off > 0; off >>= 1) {
    if (s < off) red[s] = fmaxf(red[s], red[s + off]);
    __syncthreads();
  }
  const float mx = red[0]; __syncthreads();
  const float ex = __expf(a - mx);
  red[s] = ex; __syncthreads();
  for (int off = 64; off > 0; off >>= 1) {
    if (s < off) red[s] += red[s + off];
    __syncthreads();
  }
  const float w = ex / red[0];
  sc[s] = w;
  cross_attn[((size_t)b * T_ + t) * S_ + s] = w;
  __syncthreads();

  // ctx[b,d] = sum_s w[s] * e_all[b,s,d]; each thread owns 4 d's
  float c0 = 0, c1 = 0, c2 = 0, c3 = 0;
  for (int ss = 0; ss < S_; ++ss) {
    const float* er = e_all + ((size_t)b * S_ + ss) * D_;
    const float ws = sc[ss];
    c0 += ws * er[s];       c1 += ws * er[s + 128];
    c2 += ws * er[s + 256]; c3 += ws * er[s + 384];
  }
  __bf16* g = gin + (size_t)b * KGIN;
  g[s] = (__bf16)c0;       g[s + 128] = (__bf16)c1;
  g[s + 256] = (__bf16)c2; g[s + 384] = (__bf16)c3;
  if (s < 3)             g[D_ + s] = (__bf16)x[b * 3 + s];
  if (s >= 3 && s < 32)  g[D_ + s] = (__bf16)0.0f;   // pad 515..543
}

// ---------------------------------------------------------------- GRU update
// Sums split-K partials of gi (KS_GI planes) and gh (KS_GH planes) per gate.
__global__ void __launch_bounds__(128)
gru_update(const float* __restrict__ gip, const float* __restrict__ ghp,
           float* __restrict__ h, __bf16* __restrict__ hb,
           const float* __restrict__ Wo_w, const float* __restrict__ Wo_b,
           float* __restrict__ x, float* __restrict__ outs,
           float* __restrict__ hT, int t) {
  __shared__ float red[3 * 128];
  const int b = blockIdx.x, tid = threadIdx.x;
  const size_t strideI = (size_t)B_ * N3D;
  const float* gib = gip + (size_t)b * N3D;
  const float* ghb = ghp + (size_t)b * N3D;
  float p0 = 0, p1 = 0, p2 = 0;
#pragma unroll
  for (int j = 0; j < 4; ++j) {
    const int d = tid + 128 * j;
    float gir = 0, giz = 0, gin_ = 0, ghr = 0, ghz = 0, ghn = 0;
#pragma unroll
    for (int p = 0; p < KS_GI; ++p) {
      const float* g = gib + (size_t)p * strideI;
      gir += g[d]; giz += g[D_ + d]; gin_ += g[2 * D_ + d];
    }
#pragma unroll
    for (int p = 0; p < KS_GH; ++p) {
      const float* g = ghb + (size_t)p * strideI;
      ghr += g[d]; ghz += g[D_ + d]; ghn += g[2 * D_ + d];
    }
    const float r  = 1.f / (1.f + __expf(-(gir + ghr)));
    const float z  = 1.f / (1.f + __expf(-(giz + ghz)));
    const float n  = tanhf(gin_ + r * ghn);
    const float hn = (1.f - z) * n + z * h[(size_t)b * D_ + d];
    h[(size_t)b * D_ + d]  = hn;
    hb[(size_t)b * D_ + d] = (__bf16)hn;
    if (t == T_ - 1) hT[(size_t)b * D_ + d] = hn;
    p0 += hn * Wo_w[d * 3 + 0];
    p1 += hn * Wo_w[d * 3 + 1];
    p2 += hn * Wo_w[d * 3 + 2];
  }
  red[tid] = p0; red[128 + tid] = p1; red[256 + tid] = p2;
  __syncthreads();
  for (int off = 64; off > 0; off >>= 1) {
    if (tid < off) {
      red[tid]       += red[tid + off];
      red[128 + tid] += red[128 + tid + off];
      red[256 + tid] += red[256 + tid + off];
    }
    __syncthreads();
  }
  if (tid < 3) {
    const float o = red[tid * 128] + Wo_b[tid];
    outs[((size_t)b * T_ + t) * 3 + tid] = o;
    x[b * 3 + tid] = o;
  }
}

// ---------------------------------------------------------------- launcher
extern "C" void kernel_launch(void* const* d_in, const int* in_sizes, int n_in,
                              void* d_out, int out_size, void* d_ws, size_t ws_size,
                              hipStream_t stream) {
  const float* e_all  = (const float*)d_in[0];
  const float* e_last = (const float*)d_in[1];
  const float* Wa_w   = (const float*)d_in[2];
  const float* Wa_b   = (const float*)d_in[3];
  const float* Ua_w   = (const float*)d_in[4];
  const float* Ua_b   = (const float*)d_in[5];
  const float* Va_w   = (const float*)d_in[6];
  const float* Va_b   = (const float*)d_in[7];
  const float* W_ih   = (const float*)d_in[8];
  const float* W_hh   = (const float*)d_in[9];
  const float* b_ih   = (const float*)d_in[10];
  const float* b_hh   = (const float*)d_in[11];
  const float* Wo_w   = (const float*)d_in[12];
  const float* Wo_b   = (const float*)d_in[13];

  // Workspace carve-up (~34 MB; Uk f32 dominates at 16.8 MB)
  size_t off = 0;
  auto carve = [&](size_t bytes) -> void* {
    void* p = (char*)d_ws + off;
    off += (bytes + 255) & ~(size_t)255;
    return p;
  };
  __bf16* e_bf  = (__bf16*)carve((size_t)B_ * S_ * D_ * 2);
  float*  Uk    = (float*) carve((size_t)B_ * S_ * D_ * 4);
  __bf16* Wa_p  = (__bf16*)carve((size_t)D_ * D_ * 2);
  __bf16* Ua_p  = (__bf16*)carve((size_t)D_ * D_ * 2);
  __bf16* Wih_p = (__bf16*)carve((size_t)KGIN * N3D * 2);
  __bf16* Whh_p = (__bf16*)carve((size_t)D_ * N3D * 2);
  float*  h     = (float*) carve((size_t)B_ * D_ * 4);
  __bf16* hbf   = (__bf16*)carve((size_t)B_ * D_ * 2);
  float*  qpart = (float*) carve((size_t)KS_Q * B_ * D_ * 4);
  __bf16* gin   = (__bf16*)carve((size_t)B_ * KGIN * 2);
  float*  gip   = (float*) carve((size_t)KS_GI * B_ * N3D * 4);
  float*  ghp   = (float*) carve((size_t)KS_GH * B_ * N3D * 4);
  float*  x     = (float*) carve((size_t)B_ * 3 * 4);

  // Output regions: d_outputs [B,T,3] | hT [1,B,D] | cross_attn [B,T,S]
  float* outs  = (float*)d_out;
  float* hT    = outs + (size_t)B_ * T_ * 3;
  float* cross = hT + (size_t)B_ * D_;

  // One-time precompute
  {
    int n = B_ * S_ * D_;
    cvt_f32_to_bf16<<<(n + 255) / 256, 256, 0, stream>>>(e_all, e_bf, n);
    n = D_ * D_;
    pack_b_frag<<<(n + 255) / 256, 256, 0, stream>>>(Wa_w, Wa_p, D_, D_, D_, 0);
    pack_b_frag<<<(n + 255) / 256, 256, 0, stream>>>(Ua_w, Ua_p, D_, D_, D_, 0);
    n = N3D * KGIN;
    pack_b_frag<<<(n + 255) / 256, 256, 0, stream>>>(W_ih, Wih_p, N3D, KGIN, D_ + 3, 1);
    n = N3D * D_;
    pack_b_frag<<<(n + 255) / 256, 256, 0, stream>>>(W_hh, Whh_p, N3D, D_, D_, 1);
    n = B_ * D_;
    init_state<<<(n + 255) / 256, 256, 0, stream>>>(e_last, h, hbf, x);
    // Uk = e_all @ Ua + b : [8192,512] x [512,512], KS=1 (already parallel)
    gemm_wmma_bf16<<<dim3(D_ / 64, (B_ * S_) / 64, 1), 128, 0, stream>>>(
        e_bf, Ua_p, Ua_b, Uk, B_ * S_, D_, D_, 1);
  }

  // Sequential decode: T=64 steps chained through h / x. Per-step GEMMs are
  // M=64 and latency-bound -> split-K to shorten the serial k-chain ~4x.
  for (int t = 0; t < T_; ++t) {
    // q = h @ Wa + b : [64,512] x [512,512], 4 partial planes
    gemm_wmma_bf16<<<dim3(D_ / 64, 1, KS_Q), 128, 0, stream>>>(
        hbf, Wa_p, Wa_b, qpart, B_, D_, D_, KS_Q);
    // attention: sum q partials, scores, softmax, ctx, pack GRU input
    attn_step<<<B_, 128, 0, stream>>>(Uk, qpart, e_all, Va_w, Va_b, x, cross, gin, t);
    // gi = gin @ W_ih^T + b_ih : [64,544] x [544,1536], 5 partial planes
    gemm_wmma_bf16<<<dim3(N3D / 64, 1, KS_GI), 128, 0, stream>>>(
        gin, Wih_p, b_ih, gip, B_, N3D, KGIN, KS_GI);
    // gh = h @ W_hh^T + b_hh : [64,512] x [512,1536], 4 partial planes
    gemm_wmma_bf16<<<dim3(N3D / 64, 1, KS_GH), 128, 0, stream>>>(
        hbf, Whh_p, b_hh, ghp, B_, N3D, D_, KS_GH);
    // gates (summing partials), h update, 3-wide output head
    gru_update<<<B_, 128, 0, stream>>>(gip, ghp, h, hbf, Wo_w, Wo_b, x, outs, hT, t);
  }
  (void)in_sizes; (void)n_in; (void)out_size; (void)ws_size;
}
